// HierarchicalVQVAE_61710090109548
// MI455X (gfx1250) — matrix operand order repaired and checked by reading, not compiled, not run on hardware
//
#include <hip/hip_runtime.h>
#include <hip/hip_bf16.h>

typedef __bf16 bf16_t;
typedef __attribute__((ext_vector_type(16))) __bf16 v16bf;
typedef __attribute__((ext_vector_type(8)))  float  v8f;

#define B_   4
#define T_   2048
#define D_   256
#define A_   7
#define NZ_  256
#define LN_EPS 1e-5f

union FragBF  { v16bf v; unsigned int u[8]; };
union Pack8BF { uint4 q; bf16_t h[8]; };

// ---------------------------------------------------------------------------
// 1) Convert conv weights fp32 [layer][cout][cin][k] -> bf16 [layer][k][cout][cin]
//    16 layers total: 0..7 = encoder, 8..15 = decoder.
// ---------------------------------------------------------------------------
__global__ void k_wcvt(const float* __restrict__ enc_w,
                       const float* __restrict__ dec_w,
                       bf16_t* __restrict__ wbf) {
    int tid = blockIdx.x * blockDim.x + threadIdx.x;           // over 16*3*256*256
    int ci  = tid & (D_ - 1);
    int tmp = tid >> 8;
    int co  = tmp & (D_ - 1);
    tmp >>= 8;
    int k   = tmp % 3;
    int l   = tmp / 3;
    const float* src = (l < 8) ? enc_w : dec_w;
    int ll = (l < 8) ? l : (l - 8);
    float w = src[(((size_t)ll * D_ + co) * D_ + ci) * 3 + k];
    wbf[(((size_t)l * 3 + k) * D_ + co) * D_ + ci] = (bf16_t)w;
}

// ---------------------------------------------------------------------------
// 2) Input projection: x[b,t,d] = sum_a actions[b,t,a]*W_in[a,d] + b_in[d]
// ---------------------------------------------------------------------------
__global__ void k_proj_in(const float* __restrict__ actions,
                          const float* __restrict__ W_in,
                          const float* __restrict__ b_in,
                          bf16_t* __restrict__ out) {
    int tid = blockIdx.x * blockDim.x + threadIdx.x;           // over B*T*D
    int d  = tid & (D_ - 1);
    int bt = tid >> 8;
    const float* arow = actions + (size_t)bt * A_;
    float s = b_in[d];
    #pragma unroll
    for (int a = 0; a < A_; ++a) s += arow[a] * W_in[a * D_ + d];
    out[(size_t)bt * D_ + d] = (bf16_t)s;
}

// ---------------------------------------------------------------------------
// 3) Dilated conv layer via WMMA.  Activations bf16 [B][T][D].
//    Each wave: 16(cout) x 64(t) = 4 accumulator tiles; A (weight) fragment
//    loaded once per (tap, cin-block), reused across the 4 N-tiles.
//    CHECK=false: branch-free interior path (boundary provably in-range).
// ---------------------------------------------------------------------------
template <bool CHECK>
__device__ __forceinline__ void conv_mma(const bf16_t* __restrict__ xb,
                                         const bf16_t* __restrict__ w3,
                                         int t_base, int m0, int n, int half,
                                         int dil, v8f acc[4]) {
    #pragma unroll
    for (int k = 0; k < 3; ++k) {
        const int tshift = (k - 1) * dil;
        const bf16_t* wk = w3 + (size_t)k * D_ * D_;

        for (int c0 = 0; c0 < D_; c0 += 32) {
            // A fragment: weights, row m = m0+n, K contiguous: k = 16*half + s (matches B)
            FragBF af;
            const unsigned int* wrow =
                reinterpret_cast<const unsigned int*>(wk + (size_t)(m0 + n) * D_ + c0);
            #pragma unroll
            for (int j = 0; j < 8; ++j)
                af.u[j] = wrow[half * 8 + j];

            // 4 B fragments sharing this A fragment
            #pragma unroll
            for (int s = 0; s < 4; ++s) {
                const int tsrc = t_base + s * 16 + n + tshift;
                FragBF bfg;
                if (CHECK) {
                    const bool valid = (tsrc >= 0) && (tsrc < T_);
                    unsigned int bv[8] = {0, 0, 0, 0, 0, 0, 0, 0};
                    if (valid) {
                        const unsigned int* xr = reinterpret_cast<const unsigned int*>(
                            xb + (size_t)tsrc * D_ + c0 + half * 16);
                        #pragma unroll
                        for (int j = 0; j < 8; ++j) bv[j] = xr[j];
                    }
                    #pragma unroll
                    for (int j = 0; j < 8; ++j) bfg.u[j] = bv[j];
                } else {
                    const unsigned int* xr = reinterpret_cast<const unsigned int*>(
                        xb + (size_t)tsrc * D_ + c0 + half * 16);
                    #pragma unroll
                    for (int j = 0; j < 8; ++j) bfg.u[j] = xr[j];
                }
                acc[s] = __builtin_amdgcn_wmma_f32_16x16x32_bf16(
                    false, af.v, false, bfg.v, (short)0, acc[s], false, false);
            }
        }
    }
}

__global__ __launch_bounds__(128)
void k_conv(const bf16_t* __restrict__ xin, bf16_t* __restrict__ xout,
            const bf16_t* __restrict__ w3,   // [3][D][D] bf16 for this layer
            const float*  __restrict__ bias, // [D]
            int dil) {
    const int lane   = threadIdx.x & 31;
    const int wv     = threadIdx.x >> 5;
    const int t_base = (blockIdx.x * 4 + wv) * 64;   // 64 timesteps per wave
    const int m0     = blockIdx.y * 16;
    const int b      = blockIdx.z;
    const int n      = lane & 15;
    const int half   = lane >> 4;

    const bf16_t* xb = xin + (size_t)b * T_ * D_;

    // Prefetch residual rows for the epilogue (global_prefetch_b8 path).
    __builtin_prefetch(xb + (size_t)(t_base + n) * D_ + m0, 0, 1);
    __builtin_prefetch(xb + (size_t)(t_base + 32 + n) * D_ + m0, 0, 1);

    v8f acc[4] = {{}, {}, {}, {}};

    // Wave-uniform edge test: window [t_base - dil, t_base + 63 + dil]
    if (t_base >= dil && t_base + 63 + dil < T_) {
        conv_mma<false>(xb, w3, t_base, m0, n, half, dil, acc);  // hot path
    } else {
        conv_mma<true>(xb, w3, t_base, m0, n, half, dil, acc);   // edges
    }

    // Epilogue: C/D layout — VGPR r: M = r + half*8, N = n.
    // The 8 output channels per lane are contiguous -> vector residual/store.
    const int cbase = m0 + half * 8;
    const float4* bp = reinterpret_cast<const float4*>(bias + cbase);
    const float4 bia0 = bp[0], bia1 = bp[1];
    const float biav[8] = {bia0.x, bia0.y, bia0.z, bia0.w,
                           bia1.x, bia1.y, bia1.z, bia1.w};
    bf16_t* ob = xout + (size_t)b * T_ * D_;
    #pragma unroll
    for (int s = 0; s < 4; ++s) {
        const int t = t_base + s * 16 + n;
        Pack8BF res, outp;
        res.q = *reinterpret_cast<const uint4*>(xb + (size_t)t * D_ + cbase);
        #pragma unroll
        for (int r = 0; r < 8; ++r) {
            float v = acc[s][r] + biav[r];
            v = v > 0.f ? v : 0.f;                 // ReLU
            v += (float)res.h[r];                  // identity residual
            outp.h[r] = (bf16_t)v;
        }
        *reinterpret_cast<uint4*>(ob + (size_t)t * D_ + cbase) = outp.q;
    }
}

// ---------------------------------------------------------------------------
// 4) LayerNorm over D per (b,t); one wave per row.
// ---------------------------------------------------------------------------
__global__ void k_layernorm(const bf16_t* __restrict__ xin,
                            bf16_t* __restrict__ emb,
                            const float* __restrict__ g,
                            const float* __restrict__ bln) {
    const int lane = threadIdx.x & 31;
    const int row  = blockIdx.x * (blockDim.x >> 5) + (threadIdx.x >> 5);
    const bf16_t* r = xin + (size_t)row * D_;
    float vals[8], s = 0.f, ss = 0.f;
    #pragma unroll
    for (int i = 0; i < 8; ++i) {
        float v = (float)r[lane + i * 32];
        vals[i] = v; s += v; ss += v * v;
    }
    #pragma unroll
    for (int off = 16; off; off >>= 1) {
        s  += __shfl_xor(s,  off, 32);
        ss += __shfl_xor(ss, off, 32);
    }
    const float mu  = s * (1.f / D_);
    const float var = ss * (1.f / D_) - mu * mu;
    const float inv = rsqrtf(var + LN_EPS);
    bf16_t* o = emb + (size_t)row * D_;
    #pragma unroll
    for (int i = 0; i < 8; ++i) {
        int d = lane + i * 32;
        o[d] = (bf16_t)((vals[i] - mu) * inv * g[d] + bln[d]);
    }
}

// ---------------------------------------------------------------------------
// 5) L2-normalize codebook rows; one wave per code.
// ---------------------------------------------------------------------------
__global__ void k_cbnorm(const float* __restrict__ cb, bf16_t* __restrict__ cn) {
    const int lane = threadIdx.x & 31;
    const int row  = blockIdx.x * (blockDim.x >> 5) + (threadIdx.x >> 5);
    const float* r = cb + (size_t)row * D_;
    float vals[8], ss = 0.f;
    #pragma unroll
    for (int i = 0; i < 8; ++i) { float v = r[lane + i * 32]; vals[i] = v; ss += v * v; }
    #pragma unroll
    for (int off = 16; off; off >>= 1) ss += __shfl_xor(ss, off, 32);
    const float sc = 1.f / fmaxf(sqrtf(ss), 1e-12f);
    bf16_t* o = cn + (size_t)row * D_;
    #pragma unroll
    for (int i = 0; i < 8; ++i) o[lane + i * 32] = (bf16_t)(vals[i] * sc);
}

// ---------------------------------------------------------------------------
// 6) Similarity GEMM: sim[bt, k] = emb[bt,:] . cnorm[k,:]  (WMMA, K=256)
//    Each wave: 16 tokens x 64 codes; token (A) fragment reused across the
//    4 code tiles.
// ---------------------------------------------------------------------------
__global__ __launch_bounds__(128)
void k_sim(const bf16_t* __restrict__ emb, const bf16_t* __restrict__ cn,
           float* __restrict__ sim) {
    const int lane = threadIdx.x & 31;
    const int wv   = threadIdx.x >> 5;
    const int m0   = blockIdx.x * 16;     // token rows
    const int nb   = wv * 64;             // this wave's 64 codes
    const int n    = lane & 15;
    const int half = lane >> 4;

    v8f acc[4] = {{}, {}, {}, {}};
    for (int c0 = 0; c0 < D_; c0 += 32) {
        FragBF af;
        const unsigned int* ar =
            reinterpret_cast<const unsigned int*>(emb + (size_t)(m0 + n) * D_ + c0);
        #pragma unroll
        for (int j = 0; j < 8; ++j)
            af.u[j] = ar[half * 8 + j];
        #pragma unroll
        for (int s = 0; s < 4; ++s) {
            const unsigned int* br = reinterpret_cast<const unsigned int*>(
                cn + (size_t)(nb + s * 16 + n) * D_ + c0 + half * 16);
            FragBF bfg;
            #pragma unroll
            for (int j = 0; j < 8; ++j) bfg.u[j] = br[j];
            acc[s] = __builtin_amdgcn_wmma_f32_16x16x32_bf16(
                false, af.v, false, bfg.v, (short)0, acc[s], false, false);
        }
    }
    #pragma unroll
    for (int s = 0; s < 4; ++s)
        #pragma unroll
        for (int r = 0; r < 8; ++r)
            sim[(size_t)(m0 + r + half * 8) * NZ_ + nb + s * 16 + n] = acc[s][r];
}

// ---------------------------------------------------------------------------
// 7) Argmax over codes + gather normalized code row -> decoder input.
//    (straight-through: forward value of z_st == z_q)
// ---------------------------------------------------------------------------
__global__ void k_argmax_gather(const float* __restrict__ sim,
                                const bf16_t* __restrict__ cn,
                                bf16_t* __restrict__ zq) {
    const int lane = threadIdx.x & 31;
    const int row  = blockIdx.x * (blockDim.x >> 5) + (threadIdx.x >> 5);
    const float* sr = sim + (size_t)row * NZ_;
    float bv = -3.402823e38f; int bi = 0;
    #pragma unroll
    for (int i = 0; i < 8; ++i) {
        int idx = lane + i * 32;
        float v = sr[idx];
        if (v > bv || (v == bv && idx < bi)) { bv = v; bi = idx; }
    }
    #pragma unroll
    for (int off = 16; off; off >>= 1) {
        float ov = __shfl_xor(bv, off, 32);
        int   oi = __shfl_xor(bi, off, 32);
        if (ov > bv || (ov == bv && oi < bi)) { bv = ov; bi = oi; }
    }
    // all lanes now agree on bi; copy 256 bf16 (512B) cooperatively
    const uint4* src = reinterpret_cast<const uint4*>(cn + (size_t)bi * D_);
    uint4*       dst = reinterpret_cast<uint4*>(zq + (size_t)row * D_);
    dst[lane] = src[lane];
}

// ---------------------------------------------------------------------------
// 8) Output projection: recon[bt,a] = sum_d y[bt,d]*W_out[d,a] + b_out[a]
// ---------------------------------------------------------------------------
__global__ void k_proj_out(const bf16_t* __restrict__ y,
                           const float* __restrict__ W_out,
                           const float* __restrict__ b_out,
                           float* __restrict__ out) {
    int tid = blockIdx.x * blockDim.x + threadIdx.x;    // over B*T*A
    if (tid >= B_ * T_ * A_) return;
    int a  = tid % A_;
    int bt = tid / A_;
    const bf16_t* yr = y + (size_t)bt * D_;
    float s = b_out[a];
    for (int d = 0; d < D_; ++d) s += (float)yr[d] * W_out[d * A_ + a];
    out[tid] = s;
}

// ---------------------------------------------------------------------------
extern "C" void kernel_launch(void* const* d_in, const int* in_sizes, int n_in,
                              void* d_out, int out_size, void* d_ws, size_t ws_size,
                              hipStream_t stream) {
    const float* actions    = (const float*)d_in[0];
    const float* W_in       = (const float*)d_in[1];
    const float* b_in       = (const float*)d_in[2];
    const float* enc_w      = (const float*)d_in[3];
    const float* enc_b      = (const float*)d_in[4];
    const float* ln_g       = (const float*)d_in[5];
    const float* ln_b       = (const float*)d_in[6];
    const float* codebook_z = (const float*)d_in[7];
    // d_in[8] codebook_q: unused by forward output
    const float* dec_w      = (const float*)d_in[9];
    const float* dec_b      = (const float*)d_in[10];
    const float* W_out      = (const float*)d_in[11];
    const float* b_out      = (const float*)d_in[12];
    float* recon = (float*)d_out;

    // Workspace carve-up (256B aligned)
    char* base = (char*)d_ws;
    size_t off = 0;
    auto carve = [&](size_t bytes) {
        void* p = base + off;
        off = (off + bytes + 255) & ~(size_t)255;
        return p;
    };
    bf16_t* wbf  = (bf16_t*)carve((size_t)16 * 3 * D_ * D_ * sizeof(bf16_t)); // 6.3 MB
    bf16_t* act0 = (bf16_t*)carve((size_t)B_ * T_ * D_ * sizeof(bf16_t));     // 4 MB
    bf16_t* act1 = (bf16_t*)carve((size_t)B_ * T_ * D_ * sizeof(bf16_t));     // 4 MB
    bf16_t* cn   = (bf16_t*)carve((size_t)NZ_ * D_ * sizeof(bf16_t));         // 128 KB
    float*  sim  = (float*) carve((size_t)B_ * T_ * NZ_ * sizeof(float));     // 8 MB
    (void)ws_size; (void)in_sizes; (void)n_in; (void)out_size;

    const int dils[8] = {1, 2, 4, 8, 1, 2, 4, 8};
    // block = 4 waves; each wave: 16 cout x 64 t  -> block covers 256 timesteps
    const dim3 convGrid(T_ / 256, D_ / 16, B_);

    // 1) weight convert
    k_wcvt<<<(16 * 3 * D_ * D_) / 256, 256, 0, stream>>>(enc_w, dec_w, wbf);
    // 2) input projection -> act0
    k_proj_in<<<(B_ * T_ * D_) / 256, 256, 0, stream>>>(actions, W_in, b_in, act0);

    // 3) encoder: 8 dilated conv layers, ping-pong act0/act1
    bf16_t* cur = act0; bf16_t* nxt = act1;
    for (int l = 0; l < 8; ++l) {
        k_conv<<<convGrid, 128, 0, stream>>>(cur, nxt,
            wbf + (size_t)l * 3 * D_ * D_, enc_b + (size_t)l * D_, dils[l]);
        bf16_t* t = cur; cur = nxt; nxt = t;
    }
    // cur == act0 after 8 swaps

    // 4) layernorm: act0 -> act1 (emb)
    k_layernorm<<<(B_ * T_) / 8, 256, 0, stream>>>(cur, nxt, ln_g, ln_b);
    // 5) codebook normalize, similarity GEMM, argmax+gather -> act0 (= z_q)
    k_cbnorm<<<NZ_ / 8, 256, 0, stream>>>(codebook_z, cn);
    k_sim<<<(B_ * T_) / 16, 128, 0, stream>>>(nxt, cn, sim);
    k_argmax_gather<<<(B_ * T_) / 8, 256, 0, stream>>>(sim, cn, cur);

    // 6) decoder: 8 dilated conv layers from z_q
    for (int l = 0; l < 8; ++l) {
        k_conv<<<convGrid, 128, 0, stream>>>(cur, nxt,
            wbf + (size_t)(8 + l) * 3 * D_ * D_, dec_b + (size_t)l * D_, dils[l]);
        bf16_t* t = cur; cur = nxt; nxt = t;
    }
    // cur == act0

    // 7) output projection -> recon (fp32)
    k_proj_out<<<(B_ * T_ * A_ + 255) / 256, 256, 0, stream>>>(cur, W_out, b_out, recon);
}
